// Attention_23021024706719
// MI455X (gfx1250) — matrix operand order repaired and checked by reading, compiled, not run on hardware
//
#include <hip/hip_runtime.h>
#include <hip/hip_bf16.h>

// ---------------------------------------------------------------------------
// Problem constants (from reference)
// ---------------------------------------------------------------------------
#define NE 1024   // n_embd
#define HH 16     // n_head
#define DH 64     // head dim
#define LL 1024   // seq len
#define BBATCH 2  // batch
#define RR (BBATCH * LL)  // 2048 rows for projection GEMMs
#define QF 32     // qkfeat dim

typedef __bf16 bf16_t;
typedef __attribute__((ext_vector_type(16))) __bf16 v16bf;
typedef __attribute__((ext_vector_type(8)))  float  v8f;
typedef __attribute__((ext_vector_type(4)))  int    i32x4;

// Guarded async global->LDS path (falls back to synchronous loads)
#if defined(__has_builtin)
#if __has_builtin(__builtin_amdgcn_global_load_async_to_lds_b128) && \
    __has_builtin(__builtin_amdgcn_s_wait_asynccnt)
#define ASYNC_LDS 1
#endif
#endif

union FragBF {
  v16bf v;
  uint4 u[2];
  bf16_t h[16];
};

static __device__ __forceinline__ v8f wmma_bf16(v16bf a, v16bf b, v8f c) {
  return __builtin_amdgcn_wmma_f32_16x16x32_bf16(false, a, false, b, (short)0, c,
                                                 false, false);
}

static __device__ __forceinline__ v8f zero_v8() {
  v8f z = {0.f, 0.f, 0.f, 0.f, 0.f, 0.f, 0.f, 0.f};
  return z;
}

// ---------------------------------------------------------------------------
// f32 -> bf16 conversion (elementwise, vectorized)
// ---------------------------------------------------------------------------
__global__ void cvt_f32_bf16(const float* __restrict__ in,
                             bf16_t* __restrict__ out, long n) {
  long i = ((long)blockIdx.x * blockDim.x + threadIdx.x) * 4;
  if (i + 3 < n) {
    float4 f = *(const float4*)(in + i);
    out[i + 0] = (bf16_t)f.x;
    out[i + 1] = (bf16_t)f.y;
    out[i + 2] = (bf16_t)f.z;
    out[i + 3] = (bf16_t)f.w;
  } else {
    for (; i < n; ++i) out[i] = (bf16_t)in[i];
  }
}

// ---------------------------------------------------------------------------
// Weight prep: W (K x N, f32, row-major) -> Wt (N x K, bf16)
// ---------------------------------------------------------------------------
__global__ __launch_bounds__(256) void wt_transpose_kernel(
    const float* __restrict__ W, bf16_t* __restrict__ Wt, int K, int N) {
  __shared__ float tile[32][33];
  int k0 = blockIdx.y * 32, n0 = blockIdx.x * 32;
  int tx = threadIdx.x & 31, ty = threadIdx.x >> 5;  // ty in 0..7
#pragma unroll
  for (int r = ty; r < 32; r += 8)
    tile[r][tx] = W[(size_t)(k0 + r) * N + n0 + tx];
  __syncthreads();
#pragma unroll
  for (int r = ty; r < 32; r += 8)
    Wt[(size_t)(n0 + r) * K + k0 + tx] = (bf16_t)tile[tx][r];
}

// ---------------------------------------------------------------------------
// bf16 GEMM:  C[M,N] = A[M,K] @ W[K,N] + bias[N], W given pre-transposed
// as Wt[N][K] (bf16). Block = 256 threads (8 waves), tile 64(M) x 128(N),
// wave = 32x32 -> 4 WMMAs per 32-k step with 2 A-frag + 2 B-frag reuse.
// A tile staged in LDS (async path when available); B-frags straight from
// global (weights are L2-resident).
// MODE 0: row-major OutT store.  MODE 1: bf16 store to (b,h,d,s) V-layout.
// ---------------------------------------------------------------------------
template <typename OutT, int MODE>
__global__ __launch_bounds__(256) void gemm_bf16_kernel(
    const bf16_t* __restrict__ A, const bf16_t* __restrict__ Wt,
    const float* __restrict__ bias, OutT* __restrict__ Cmat,
    int M, int N, int K) {
  __shared__ bf16_t As[64][32];

  int tid = threadIdx.x;
  int wave = tid >> 5, lane = tid & 31;
  int lh = lane & 15, hi = lane >> 4;
  int bm = blockIdx.y * 64, bn = blockIdx.x * 128;
  int wm = (wave >> 2) * 32, wn = (wave & 3) * 32;
  int kg = hi * 8, ks = hi * 16;

  v8f acc[2][2];
#pragma unroll
  for (int i = 0; i < 2; ++i)
#pragma unroll
    for (int j = 0; j < 2; ++j) acc[i][j] = zero_v8();

  const bf16_t* bptr0 = Wt + (size_t)(bn + wn + lh) * K;
  const bf16_t* bptr1 = Wt + (size_t)(bn + wn + 16 + lh) * K;

  int r = tid >> 2, c = (tid & 3) * 8;
  for (int k0 = 0; k0 < K; k0 += 32) {
    const bf16_t* src = A + (size_t)(bm + r) * K + k0 + c;
#ifdef ASYNC_LDS
    __builtin_amdgcn_global_load_async_to_lds_b128(
        (i32x4*)(void*)src, (i32x4*)(void*)&As[r][c], 0, 0);
#else
    *(uint4*)&As[r][c] = *(const uint4*)src;
#endif
    if (k0 + 32 < K) __builtin_prefetch(src + 32, 0, 0);
#ifdef ASYNC_LDS
    __builtin_amdgcn_s_wait_asynccnt(0);
#endif
    __syncthreads();

    FragBF a0, a1, b0, b1;
    a0.u[0] = *(const uint4*)&As[wm + lh][ks];
    a0.u[1] = *(const uint4*)&As[wm + lh][ks + 8];
    a1.u[0] = *(const uint4*)&As[wm + 16 + lh][ks];
    a1.u[1] = *(const uint4*)&As[wm + 16 + lh][ks + 8];
    b0.u[0] = *(const uint4*)(bptr0 + k0 + ks);
    b0.u[1] = *(const uint4*)(bptr0 + k0 + ks + 8);
    b1.u[0] = *(const uint4*)(bptr1 + k0 + ks);
    b1.u[1] = *(const uint4*)(bptr1 + k0 + ks + 8);

    acc[0][0] = wmma_bf16(a0.v, b0.v, acc[0][0]);
    acc[0][1] = wmma_bf16(a0.v, b1.v, acc[0][1]);
    acc[1][0] = wmma_bf16(a1.v, b0.v, acc[1][0]);
    acc[1][1] = wmma_bf16(a1.v, b1.v, acc[1][1]);
    __syncthreads();
  }

#pragma unroll
  for (int ni = 0; ni < 2; ++ni) {
    int n = bn + wn + ni * 16 + lh;
    float bsc = bias ? bias[n] : 0.f;
#pragma unroll
    for (int mi = 0; mi < 2; ++mi) {
#pragma unroll
      for (int rr2 = 0; rr2 < 8; ++rr2) {
        int m = bm + wm + mi * 16 + rr2 + hi * 8;
        float val = acc[mi][ni][rr2] + bsc;
        if (MODE == 0) {
          Cmat[(size_t)m * N + n] = (OutT)val;
        } else {
          // V transposed store: (b, h, d, s)
          int bb = m >> 10;          // m / LL
          int ss = m & (LL - 1);
          int hh2 = n >> 6;          // n / DH
          int dd = n & (DH - 1);
          Cmat[(((size_t)bb * HH + hh2) * DH + dd) * LL + ss] = (OutT)val;
        }
      }
    }
  }
}

// ---------------------------------------------------------------------------
// qkfeat bias: bias16[row, h] = qkfeat[row, 0:32] @ Wf[:, h] + bf[h]
// rows = B*L*S (2M). One 16x16x32 WMMA per 16 rows. Output bf16 (B,L,S,H).
// ---------------------------------------------------------------------------
__global__ __launch_bounds__(256) void qkbias_kernel(
    const float* __restrict__ qkfeat, const float* __restrict__ Wf,
    const float* __restrict__ bfv, bf16_t* __restrict__ out) {
  int tid = threadIdx.x;
  int wave = tid >> 5, lane = tid & 31;
  int lh = lane & 15, hi = lane >> 4;
  long r0 = ((long)blockIdx.x * 8 + wave) * 16;

  FragBF bfr;
  int ks = hi * 16;
#pragma unroll
  for (int e = 0; e < 16; ++e) bfr.h[e] = (bf16_t)Wf[(ks + e) * HH + lh];

  const float* row = qkfeat + (r0 + lh) * QF;
  int kg = hi * 8;
  FragBF a;
#pragma unroll
  for (int e = 0; e < 8; ++e) a.h[e] = (bf16_t)row[ks + e];
#pragma unroll
  for (int e = 0; e < 8; ++e) a.h[8 + e] = (bf16_t)row[ks + 8 + e];

  v8f c = zero_v8();
  c = wmma_bf16(a.v, bfr.v, c);

  float bb = bfv[lh];
#pragma unroll
  for (int r = 0; r < 8; ++r) {
    long m = r0 + r + hi * 8;
    out[m * HH + lh] = (bf16_t)(c[r] + bb);
  }
}

// ---------------------------------------------------------------------------
// Flash attention, one (b, h, 64-row l tile) per block, 4 waves x 16 rows.
// Loop over 32-wide s tiles with causal early-exit; online softmax in f32.
// V comes pre-transposed (b,h,d,s) so P@V B-frags are contiguous global loads.
// ---------------------------------------------------------------------------
__global__ __launch_bounds__(128) void flash_kernel(
    const bf16_t* __restrict__ qm, const bf16_t* __restrict__ km,
    const bf16_t* __restrict__ vt, const bf16_t* __restrict__ biasm,
    const unsigned char* __restrict__ kmask, bf16_t* __restrict__ ym) {
  __shared__ bf16_t Pt[4][16][32];  // per-wave P tile (l x s)

  int tid = threadIdx.x, wave = tid >> 5, lane = tid & 31;
  int lh = lane & 15, hi = lane >> 4;
  int b = blockIdx.z;
  int h = blockIdx.y;
  int l0 = blockIdx.x * 64;
  int lw = l0 + wave * 16;
  int kg = hi * 8, ks = hi * 16;

  // q fragments for K slices 0..31 and 32..63 (loaded once)
  FragBF qa[2];
  const bf16_t* qrow = qm + (size_t)(b * LL + lw + lh) * NE + h * DH;
#pragma unroll
  for (int s = 0; s < 2; ++s) {
    qa[s].u[0] = *(const uint4*)(qrow + s * 32 + ks);
    qa[s].u[1] = *(const uint4*)(qrow + s * 32 + ks + 8);
  }

  const bf16_t* vbase = vt + (size_t)(b * HH + h) * DH * LL;

  v8f yacc[4];
  float mrow[8], lrow[8];
#pragma unroll
  for (int d = 0; d < 4; ++d) yacc[d] = zero_v8();
#pragma unroll
  for (int r = 0; r < 8; ++r) { mrow[r] = -3.0e38f; lrow[r] = 0.f; }

  int smax = l0 + 64;  // causal: no key index beyond last row of this tile
  for (int s0 = 0; s0 < smax; s0 += 32) {
    // scores: S = q . k^T  (two 16-col subtiles, K = 64 in two slices)
    v8f sc[2];
#pragma unroll
    for (int j = 0; j < 2; ++j) {
      v8f c = zero_v8();
#pragma unroll
      for (int s = 0; s < 2; ++s) {
        FragBF kb;
        const bf16_t* kp = km + (size_t)(b * LL + s0 + j * 16 + lh) * NE +
                           h * DH + s * 32 + ks;
        kb.u[0] = *(const uint4*)kp;
        kb.u[1] = *(const uint4*)(kp + 8);
        c = wmma_bf16(qa[s].v, kb.v, c);
      }
      sc[j] = c;
    }

    // bias + scale + causal/key masking
#pragma unroll
    for (int j = 0; j < 2; ++j) {
      int sidx = s0 + j * 16 + lh;
      bool kok = kmask[b * LL + sidx] != 0;
      const bf16_t* bp = biasm + ((size_t)(b * LL + lw) * LL + sidx) * HH + h;
#pragma unroll
      for (int r = 0; r < 8; ++r) {
        int li = lw + r + hi * 8;
        float bias = (float)bp[(size_t)(r + hi * 8) * LL * HH];
        float val = (sc[j][r] + bias) * 0.125f;  // 1/sqrt(64)
        if (sidx > li || !kok) val = -3.0e38f;
        sc[j][r] = val;
      }
    }

    // online softmax update (row reduction across the 16-lane half)
#pragma unroll
    for (int r = 0; r < 8; ++r) {
      float mx = fmaxf(sc[0][r], sc[1][r]);
#pragma unroll
      for (int o = 1; o < 16; o <<= 1) mx = fmaxf(mx, __shfl_xor(mx, o, 32));
      float mnew = fmaxf(mrow[r], mx);
      float corr = __expf(mrow[r] - mnew);
      float p0 = __expf(sc[0][r] - mnew);
      float p1 = __expf(sc[1][r] - mnew);
      mrow[r] = mnew;
      float rs = p0 + p1;
#pragma unroll
      for (int o = 1; o < 16; o <<= 1) rs += __shfl_xor(rs, o, 32);
      lrow[r] = lrow[r] * corr + rs;
      sc[0][r] = p0;
      sc[1][r] = p1;
#pragma unroll
      for (int d = 0; d < 4; ++d) yacc[d][r] *= corr;
    }

    // P -> LDS (bf16, A-matrix layout source)
#pragma unroll
    for (int j = 0; j < 2; ++j)
#pragma unroll
      for (int r = 0; r < 8; ++r)
        Pt[wave][r + hi * 8][j * 16 + lh] = (bf16_t)sc[j][r];

    __syncthreads();  // Pt visible

    // y += P @ V  (4 d-subtiles; V pre-transposed -> contiguous B-frags)
    FragBF pa;
    pa.u[0] = *(const uint4*)&Pt[wave][lh][ks];
    pa.u[1] = *(const uint4*)&Pt[wave][lh][ks + 8];
#pragma unroll
    for (int d = 0; d < 4; ++d) {
      FragBF vb;
      const bf16_t* vp = vbase + (size_t)(d * 16 + lh) * LL + s0 + ks;
      vb.u[0] = *(const uint4*)vp;
      vb.u[1] = *(const uint4*)(vp + 8);
      yacc[d] = wmma_bf16(pa.v, vb.v, yacc[d]);
    }
    __syncthreads();  // protect Pt before next iteration's writes
  }

  // normalize and write y (bf16, (B, L, H*DH) row-major)
#pragma unroll
  for (int r = 0; r < 8; ++r) {
    float inv = 1.f / fmaxf(lrow[r], 1e-30f);
    int li = lw + r + hi * 8;
#pragma unroll
    for (int d = 0; d < 4; ++d) {
      ym[(size_t)(b * LL + li) * NE + h * DH + d * 16 + lh] =
          (bf16_t)(yacc[d][r] * inv);
    }
  }
}

// ---------------------------------------------------------------------------
// Launch
// ---------------------------------------------------------------------------
extern "C" void kernel_launch(void* const* d_in, const int* in_sizes, int n_in,
                              void* d_out, int out_size, void* d_ws,
                              size_t ws_size, hipStream_t stream) {
  (void)in_sizes; (void)n_in; (void)out_size; (void)ws_size;

  const float* query  = (const float*)d_in[0];
  const float* key    = (const float*)d_in[1];
  const float* value  = (const float*)d_in[2];
  const float* qkfeat = (const float*)d_in[3];
  const unsigned char* kmask = (const unsigned char*)d_in[4];
  const float* Wq = (const float*)d_in[5];
  const float* bq = (const float*)d_in[6];
  const float* Wk = (const float*)d_in[7];
  const float* bk = (const float*)d_in[8];
  const float* Wv = (const float*)d_in[9];
  const float* bv = (const float*)d_in[10];
  const float* Wf = (const float*)d_in[11];
  const float* bfv = (const float*)d_in[12];
  const float* Wp = (const float*)d_in[13];
  const float* bp = (const float*)d_in[14];
  float* out = (float*)d_out;

  char* w = (char*)d_ws;
  size_t o = 0;
  auto carve = [&](size_t bytes) -> void* {
    void* p = w + o;
    o += (bytes + 255) & ~(size_t)255;
    return p;
  };
  bf16_t* wqT = (bf16_t*)carve((size_t)NE * NE * 2);  // transposed bf16 weights
  bf16_t* wkT = (bf16_t*)carve((size_t)NE * NE * 2);
  bf16_t* wvT = (bf16_t*)carve((size_t)NE * NE * 2);
  bf16_t* wpT = (bf16_t*)carve((size_t)NE * NE * 2);
  bf16_t* xq_b = (bf16_t*)carve((size_t)RR * NE * 2);
  bf16_t* xk_b = (bf16_t*)carve((size_t)RR * NE * 2);
  bf16_t* xv_b = (bf16_t*)carve((size_t)RR * NE * 2);
  bf16_t* q_b  = (bf16_t*)carve((size_t)RR * NE * 2);
  bf16_t* k_b  = (bf16_t*)carve((size_t)RR * NE * 2);
  bf16_t* vt_b = (bf16_t*)carve((size_t)RR * NE * 2);  // (b,h,d,s)
  bf16_t* y_b  = (bf16_t*)carve((size_t)RR * NE * 2);
  bf16_t* bias_b = (bf16_t*)carve((size_t)BBATCH * LL * LL * HH * 2);  // 64MB

  dim3 tg(NE / 32, NE / 32);
  wt_transpose_kernel<<<tg, 256, 0, stream>>>(Wq, wqT, NE, NE);
  wt_transpose_kernel<<<tg, 256, 0, stream>>>(Wk, wkT, NE, NE);
  wt_transpose_kernel<<<tg, 256, 0, stream>>>(Wv, wvT, NE, NE);
  wt_transpose_kernel<<<tg, 256, 0, stream>>>(Wp, wpT, NE, NE);

  auto cvt = [&](const float* src, bf16_t* dst, long n) {
    unsigned blocks = (unsigned)((n / 4 + 255) / 256);
    cvt_f32_bf16<<<dim3(blocks), dim3(256), 0, stream>>>(src, dst, n);
  };
  cvt(query, xq_b, (long)RR * NE);
  cvt(key, xk_b, (long)RR * NE);
  cvt(value, xv_b, (long)RR * NE);

  dim3 gg(NE / 128, RR / 64);
  gemm_bf16_kernel<bf16_t, 0><<<gg, 256, 0, stream>>>(xq_b, wqT, bq, q_b, RR, NE, NE);
  gemm_bf16_kernel<bf16_t, 0><<<gg, 256, 0, stream>>>(xk_b, wkT, bk, k_b, RR, NE, NE);
  gemm_bf16_kernel<bf16_t, 1><<<gg, 256, 0, stream>>>(xv_b, wvT, bv, vt_b, RR, NE, NE);

  long nrows = (long)BBATCH * LL * LL;  // 2M rows of 32 features
  qkbias_kernel<<<dim3((unsigned)(nrows / 128)), 256, 0, stream>>>(qkfeat, Wf,
                                                                   bfv, bias_b);

  flash_kernel<<<dim3(LL / 64, HH, BBATCH), 128, 0, stream>>>(
      q_b, k_b, vt_b, bias_b, kmask, y_b);

  gemm_bf16_kernel<float, 0><<<gg, 256, 0, stream>>>(y_b, wpT, bp, out, RR, NE, NE);
}
